// MoELayer_3582002725417
// MI455X (gfx1250) — matrix operand order repaired and checked by reading, not compiled, not run on hardware
//
#include <hip/hip_runtime.h>

// Problem constants (match reference)
#define E_     8
#define DIN    128
#define DHID   256
#define DOUT   128
#define MTILE  128     // batch rows per workgroup (8 waves x 16 rows)

// Padded LDS row strides in BYTES (multiple of 16 for b128 alignment,
// ≡16 mod 256 so the 16 rows/cols a wave touches per fragment avoid bank camp)
#define XSTR   272     // 128 bf16 (256B) + 16B pad
#define HSTR   528     // 256 bf16 (512B) + 16B pad
#define W1STR  272
#define W2STR  528

#define OFF_X   0
#define OFF_H   (OFF_X  + MTILE * XSTR)     //  34816
#define OFF_W1  (OFF_H  + MTILE * HSTR)     // 102400
#define OFF_W2  (OFF_W1 + DHID  * W1STR)    // 172032
#define OFF_B1  (OFF_W2 + DOUT  * W2STR)    // 239616
#define OFF_B2  (OFF_B1 + DHID * 4)         // 240640
#define OFF_WT  (OFF_B2 + DOUT * 4)         // 241152
#define SMEM_BYTES (OFF_WT + MTILE * E_ * 4) // 245248 (< 320KB WGP LDS)

typedef __attribute__((ext_vector_type(16))) __bf16 v16bf;
typedef __attribute__((ext_vector_type(4)))  __bf16 v4bf;
typedef __attribute__((ext_vector_type(8)))  float  v8f;

union Frag { v16bf v; uint4 q[2]; };

static __device__ __forceinline__ unsigned short bf16_bits(float f) {
  __bf16 h = (__bf16)f;                       // native v_cvt (RNE) on gfx1250
  return __builtin_bit_cast(unsigned short, h);
}

// ---------------------------------------------------------------------------
// Prep: convert W1 [E][DIN][DHID] f32 -> W1T [E][DHID][DIN] bf16
//       convert W2 [E][DHID][DOUT] f32 -> W2T [E][DOUT][DHID] bf16
// (transposed so WMMA B-fragments are K-contiguous per output column)
// ---------------------------------------------------------------------------
__global__ void moe_convert_weights(const float* __restrict__ W1,
                                    const float* __restrict__ W2,
                                    unsigned short* __restrict__ W1T,
                                    unsigned short* __restrict__ W2T) {
  int idx = blockIdx.x * blockDim.x + threadIdx.x;   // 0 .. 8*128*256-1
  int e   = idx / (DIN * DHID);
  int rem = idx % (DIN * DHID);
  { // W1
    int i = rem / DHID, h = rem % DHID;
    W1T[(size_t)e * DHID * DIN + h * DIN + i] = bf16_bits(W1[idx]);
  }
  { // W2 (same flat element count)
    int h = rem / DOUT, o = rem % DOUT;
    W2T[(size_t)e * DOUT * DHID + o * DHID + h] = bf16_bits(W2[idx]);
  }
}

// ---------------------------------------------------------------------------
// Main fused MoE kernel. One workgroup = 128-row batch tile, 8 waves,
// each wave owns 16 rows end-to-end (GEMM1 -> ReLU -> GEMM2 -> combine).
// ---------------------------------------------------------------------------
extern __shared__ char smem[];

__global__ __launch_bounds__(256, 1)
void moe_wmma_kernel(const float* __restrict__ x,
                     const float* __restrict__ wts,   // [B][E]
                     const unsigned short* __restrict__ W1T,
                     const unsigned short* __restrict__ W2T,
                     const float* __restrict__ b1,    // [E][DHID]
                     const float* __restrict__ b2,    // [E][DOUT]
                     float* __restrict__ out)         // [B][DOUT]
{
  char* sm = smem;
  const int tid   = threadIdx.x;
  const int lane  = tid & 31;
  const int wave  = tid >> 5;              // 0..7
  const int l15   = lane & 15;
  const int khalf = lane >> 4;             // 0: K-low half, 1: K-high half
  const int r0    = blockIdx.x * MTILE;    // global batch row base

  // --- stage x tile (f32 -> bf16, native cvt) into LDS, padded rows ---
  {
    const float4* xs = (const float4*)(x + (size_t)r0 * DIN); // 4096 float4
    for (int c = tid; c < 4096; c += 256) {
      int row = c >> 5, seg = c & 31;
      float4 f = xs[c];
      v4bf u;
      u[0] = (__bf16)f.x; u[1] = (__bf16)f.y;
      u[2] = (__bf16)f.z; u[3] = (__bf16)f.w;
      *(v4bf*)(sm + OFF_X + row * XSTR + seg * 8) = u;
    }
    // stage routing-weights tile [128][8] f32
    const float* wsrc = wts + (size_t)r0 * E_;
    float* lwt = (float*)(sm + OFF_WT);
    for (int c = tid; c < MTILE * E_; c += 256) lwt[c] = wsrc[c];
  }
  __syncthreads();

  // --- per-lane fragment base addresses (contiguous K: lane half h holds k 16h..16h+15) ---
  const int mrow = wave * 16 + l15;                      // A-frag row (this wave)
  const int xA   = OFF_X + mrow * XSTR + khalf * 32;     // x A-frag base
  const int hA   = OFF_H + mrow * HSTR + khalf * 32;     // h A-frag base
  const int wrow_lo = wave * 16 + ((lane < 16) ? 0 : 8); // C-layout row of acc[0]
  const int hS   = OFF_H + wrow_lo * HSTR + l15 * 2;     // h store base

  // hoist x A-fragments: reused for all 16 n-tiles of every expert
  Frag ax[4];
#pragma unroll
  for (int kb = 0; kb < 4; ++kb) {
    ax[kb].q[0] = *(const uint4*)(sm + xA + kb * 64);
    ax[kb].q[1] = *(const uint4*)(sm + xA + kb * 64 + 16);
  }

  // persistent weighted-output accumulators: 8 n-tiles x 8 rows = 64 VGPRs
  v8f oacc[8];
#pragma unroll
  for (int ob = 0; ob < 8; ++ob)
#pragma unroll
    for (int j = 0; j < 8; ++j) oacc[ob][j] = 0.0f;

  for (int e = 0; e < E_; ++e) {
    __syncthreads();   // previous expert done reading W LDS buffers

    // --- stage this expert's W1T/W2T (bf16) + biases into LDS ---
    {
      const uint4* w1s = (const uint4*)(W1T + (size_t)e * DHID * DIN);  // 4096
      for (int c = tid; c < 4096; c += 256) {
        int row = c >> 4, seg = c & 15;                  // 256 rows x 16 uint4
        *(uint4*)(sm + OFF_W1 + row * W1STR + seg * 16) = w1s[c];
      }
      const uint4* w2s = (const uint4*)(W2T + (size_t)e * DOUT * DHID); // 4096
      for (int c = tid; c < 4096; c += 256) {
        int row = c >> 5, seg = c & 31;                  // 128 rows x 32 uint4
        *(uint4*)(sm + OFF_W2 + row * W2STR + seg * 16) = w2s[c];
      }
      if (tid < DHID) ((float*)(sm + OFF_B1))[tid] = b1[(size_t)e * DHID + tid];
      if (tid < DOUT) ((float*)(sm + OFF_B2))[tid] = b2[(size_t)e * DOUT + tid];
    }
    __syncthreads();

    // prefetch next expert's weights (L2 -> WGP cache) behind the WMMA work
    if (e + 1 < E_) {
      const char* p1 = (const char*)(W1T + (size_t)(e + 1) * DHID * DIN);
      const char* p2 = (const char*)(W2T + (size_t)(e + 1) * DOUT * DHID);
      // 64KB each; 256B cachelines -> 256 lines each; 1 line/thread/buffer
      __builtin_prefetch(p1 + tid * 256, 0, 1);
      __builtin_prefetch(p2 + tid * 256, 0, 1);
    }

    // --- GEMM1: h[16 x 256] = relu(x[16 x 128] @ W1 + b1), wave-private ---
    for (int nb = 0; nb < 16; ++nb) {
      v8f acc;
#pragma unroll
      for (int j = 0; j < 8; ++j) acc[j] = 0.0f;
      const int wB = OFF_W1 + (nb * 16 + l15) * W1STR + khalf * 32;
#pragma unroll
      for (int kb = 0; kb < 4; ++kb) {
        Frag b;
        b.q[0] = *(const uint4*)(sm + wB + kb * 64);
        b.q[1] = *(const uint4*)(sm + wB + kb * 64 + 16);
        acc = __builtin_amdgcn_wmma_f32_16x16x32_bf16(
            false, ax[kb].v, false, b.v, (short)0, acc, false, false);
      }
      const float bias = ((const float*)(sm + OFF_B1))[nb * 16 + l15];
#pragma unroll
      for (int j = 0; j < 8; ++j) {
        float v = fmaxf(acc[j] + bias, 0.0f);            // v_max_num_f32
        *(__bf16*)(sm + hS + j * HSTR + nb * 32) = (__bf16)v;
      }
    }
    // h rows are wave-private; same-wave DS ordering (DScnt) suffices.

    // routing weights for this wave's 16 rows, expert e
    float wj[8];
#pragma unroll
    for (int j = 0; j < 8; ++j)
      wj[j] = ((const float*)(sm + OFF_WT))[(wrow_lo + j) * E_ + e];

    // --- GEMM2: y[16 x 128] = h @ W2 + b2; weighted combine into oacc ---
    // process output n-tiles in pairs sharing one A-fragment load
    for (int obp = 0; obp < 4; ++obp) {
      const int ob0 = obp * 2, ob1 = ob0 + 1;
      v8f acc0, acc1;
#pragma unroll
      for (int j = 0; j < 8; ++j) { acc0[j] = 0.0f; acc1[j] = 0.0f; }
      const int wB2a = OFF_W2 + (ob0 * 16 + l15) * W2STR + khalf * 32;
      const int wB2b = wB2a + 16 * W2STR;
#pragma unroll
      for (int kb = 0; kb < 8; ++kb) {
        Frag a, b0, b1f;
        a.q[0]   = *(const uint4*)(sm + hA + kb * 64);
        a.q[1]   = *(const uint4*)(sm + hA + kb * 64 + 16);
        b0.q[0]  = *(const uint4*)(sm + wB2a + kb * 64);
        b0.q[1]  = *(const uint4*)(sm + wB2a + kb * 64 + 16);
        b1f.q[0] = *(const uint4*)(sm + wB2b + kb * 64);
        b1f.q[1] = *(const uint4*)(sm + wB2b + kb * 64 + 16);
        acc0 = __builtin_amdgcn_wmma_f32_16x16x32_bf16(
            false, a.v, false, b0.v, (short)0, acc0, false, false);
        acc1 = __builtin_amdgcn_wmma_f32_16x16x32_bf16(
            false, a.v, false, b1f.v, (short)0, acc1, false, false);
      }
      const float bias0 = ((const float*)(sm + OFF_B2))[ob0 * 16 + l15];
      const float bias1 = ((const float*)(sm + OFF_B2))[ob1 * 16 + l15];
#pragma unroll
      for (int j = 0; j < 8; ++j) {
        oacc[ob0][j] = fmaf(wj[j], acc0[j] + bias0, oacc[ob0][j]);
        oacc[ob1][j] = fmaf(wj[j], acc1[j] + bias1, oacc[ob1][j]);
      }
    }
  }

  // --- epilogue: store combined output (f32) ---
#pragma unroll
  for (int ob = 0; ob < 8; ++ob)
#pragma unroll
    for (int j = 0; j < 8; ++j)
      out[(size_t)(r0 + wrow_lo + j) * DOUT + ob * 16 + l15] = oacc[ob][j];
}

// ---------------------------------------------------------------------------
extern "C" void kernel_launch(void* const* d_in, const int* in_sizes, int n_in,
                              void* d_out, int out_size, void* d_ws, size_t ws_size,
                              hipStream_t stream) {
  const float* x  = (const float*)d_in[0];
  const float* wt = (const float*)d_in[1];
  const float* W1 = (const float*)d_in[2];
  const float* b1 = (const float*)d_in[3];
  const float* W2 = (const float*)d_in[4];
  const float* b2 = (const float*)d_in[5];
  float* out = (float*)d_out;

  unsigned short* W1T = (unsigned short*)d_ws;                 // 8*256*128 bf16
  unsigned short* W2T = W1T + (size_t)E_ * DHID * DIN;         // 8*128*256 bf16

  // 262144 elements each, handled by the same index space
  moe_convert_weights<<<(E_ * DIN * DHID) / 256, 256, 0, stream>>>(W1, W2, W1T, W2T);

  (void)hipFuncSetAttribute((const void*)moe_wmma_kernel,
                            hipFuncAttributeMaxDynamicSharedMemorySize,
                            SMEM_BYTES);

  const int B = in_sizes[0] / DIN;   // 65536
  moe_wmma_kernel<<<B / MTILE, 256, SMEM_BYTES, stream>>>(x, wt, W1T, W2T, b1, b2, out);
}
